// ConcatenationAttention_3796751089826
// MI455X (gfx1250) — matrix operand order repaired and checked by reading, not compiled, not run on hardware
//
#include <hip/hip_runtime.h>

typedef __attribute__((ext_vector_type(16))) _Float16 v16h;
typedef __attribute__((ext_vector_type(8)))  float    v8f;

#define B_ 8
#define S_ 1024
#define H_ 128
#define A_ 16

// ---------------- fast math helpers (gfx1250 TRANS ops) ----------------
__device__ __forceinline__ float fast_rcp(float x) {
#if __has_builtin(__builtin_amdgcn_rcpf)
  return __builtin_amdgcn_rcpf(x);          // v_rcp_f32
#else
  return 1.0f / x;
#endif
}

__device__ __forceinline__ float fast_exp2(float x) {
#if __has_builtin(__builtin_amdgcn_exp2f)
  return __builtin_amdgcn_exp2f(x);         // v_exp_f32 (2^x semantics)
#else
  return exp2f(x);
#endif
}

__device__ __forceinline__ float fast_tanh(float v) {
#if __has_builtin(__builtin_amdgcn_tanhf)
  return __builtin_amdgcn_tanhf(v);         // v_tanh_f32 (gfx1250 native)
#else
  // tanh(x) = 1 - 2/(2^(2*log2(e)*x)+1); saturates correctly at +-1
  float e = fast_exp2(v * 2.885390081777927f);   // 2*log2(e)
  return 1.0f - 2.0f * fast_rcp(e + 1.0f);
#endif
}

__device__ __forceinline__ float fast_exp(float x) {
  return fast_exp2(x * 1.4426950408889634f);     // log2(e)
}

// ---------------- Phase 1: F = x @ W1[:H],  G = x @ W1[H:] + b1 ----------
// One thread per (row, a). rows = B*S = 8192, a in [0,16). 128-MAC dot each.
// b1 is folded into G so the score loop never touches it.
__global__ __launch_bounds__(256) void fg_kernel(const float* __restrict__ x,
                                                 const float* __restrict__ W1,
                                                 const float* __restrict__ b1,
                                                 float* __restrict__ F,
                                                 float* __restrict__ G) {
  int idx = blockIdx.x * 256 + threadIdx.x;   // 0 .. B*S*A-1
  int row = idx >> 4;
  int a   = idx & 15;
  const float* xr = x + row * H_;
  float f = 0.f, g = 0.f;
#pragma unroll 8
  for (int h = 0; h < H_; ++h) {
    float xv = xr[h];
    f += xv * W1[h * A_ + a];
    g += xv * W1[(H_ + h) * A_ + a];
  }
  F[row * A_ + a] = f;
  G[row * A_ + a] = g + b1[a];
}

// ---------------- Phase 2: fused masked softmax-attention ----------------
// One workgroup (256 thr = 8 waves) per (b, i-tile of 16 rows).
// Stream j in chunks of 32: score tile -> exp tile (LDS, f16) -> WMMA PV.
__global__ __launch_bounds__(256) void attn_kernel(const float* __restrict__ x,
                                                   const float* __restrict__ F,
                                                   const float* __restrict__ G,
                                                   const float* __restrict__ w2,
                                                   const float* __restrict__ b2p,
                                                   float* __restrict__ out) {
  __shared__ float sG[32 * 16];                       // g(+b1) rows of this j-chunk
  __shared__ float sDen[16];                          // running softmax denom
  __shared__ __attribute__((aligned(16))) _Float16 sEh[16 * 32]; // e tile [ii][jj] f16
  __shared__ float sEf[32 * 16];                      // e tile [jj][ii] f32 (denom)

  const int tid = threadIdx.x;
  const int b   = blockIdx.x >> 6;
  // Reverse i-tile order: heaviest (most j-chunks) tiles launch first so the
  // triangular imbalance fills the scheduling tail with light tiles.
  const int i0  = (63 - (blockIdx.x & 63)) << 4;

  if (tid < 16) sDen[tid] = 0.f;
  const float b2v = b2p[0];

  const int lane = tid & 31;
  const int wv   = tid >> 5;        // wave id 0..7 -> h tile
  const int hi   = lane >> 4;       // lane half
  const int lm   = lane & 15;
  const int colH = wv * 16 + lm;    // this lane's h column (B/N index)

  const int ii0 = (2 * tid) & 15;   // score pair rows handled by this thread
  const int ii1 = ii0 + 1;
  const int jj  = tid >> 3;         // 0..31

  // Loop-invariant per thread: the two f rows and the w2 vector in registers.
  // w2/b2 have uniform addresses -> scalarized to s_loads by the compiler.
  float fr0[16], fr1[16], wr[16];
#pragma unroll
  for (int a = 0; a < 16; ++a) {
    fr0[a] = F[(b * S_ + i0 + ii0) * A_ + a];
    fr1[a] = F[(b * S_ + i0 + ii1) * A_ + a];
    wr[a]  = w2[a];
  }

  v8f acc = {0.f, 0.f, 0.f, 0.f, 0.f, 0.f, 0.f, 0.f};

  const int jmax = i0 + 14;         // largest j ever needed (j < i, i <= i0+15)
  const int nch  = jmax / 32 + 1;

  for (int c = 0; c < nch; ++c) {
    const int j0 = c * 32;
    __syncthreads();                // protect sEh/sG from previous iteration readers

    // stage g tile [32][16]: 512 floats, 2 per thread, coalesced
    sG[tid]       = G[(b * S_ + j0 +      (tid >> 4)) * A_ + (tid & 15)];
    sG[tid + 256] = G[(b * S_ + j0 + 16 + (tid >> 4)) * A_ + (tid & 15)];
    __syncthreads();

    // scores for pairs (ii0,jj) and (ii1,jj): g row already includes b1
    float s0 = b2v, s1 = b2v;
#pragma unroll
    for (int a = 0; a < 16; ++a) {
      float gv = sG[jj * 16 + a];
      s0 += wr[a] * fast_tanh(fr0[a] + gv);
      s1 += wr[a] * fast_tanh(fr1[a] + gv);
    }
    const int j = j0 + jj;
    float e0 = (j < i0 + ii0) ? fast_exp(s0) : 0.f;   // strictly-lower mask (j < i)
    float e1 = (j < i0 + ii1) ? fast_exp(s1) : 0.f;
    sEh[ii0 * 32 + jj] = (_Float16)e0;
    sEh[ii1 * 32 + jj] = (_Float16)e1;
    sEf[jj * 16 + ii0] = e0;
    sEf[jj * 16 + ii1] = e1;
    __syncthreads();

    // accumulate denominator (one thread per i-row)
    if (tid < 16) {
      float d = 0.f;
#pragma unroll 8
      for (int q = 0; q < 32; ++q) d += sEf[q * 16 + tid];
      sDen[tid] += d;
    }

    // A operand: 16x32 f16 e-tile. Lane: M=lm; ISA K packing per VGPR pair.
    union { v16h v; unsigned int u[8]; } Am;
#pragma unroll
    for (int p = 0; p < 8; ++p) {
      const int K0 = ((2 * p) & 7) + ((p >= 4) ? 16 : 0) + 8 * hi; // even
      Am.u[p] = *(const unsigned int*)&sEh[lm * 32 + K0];
    }

    // B operand: 32x16 f16 x-tile; lane column colH, K = (t&7) + 16*(t>>3) + 8*hi.
    v16h Bm;
#pragma unroll
    for (int t = 0; t < 16; ++t) {
      const int K = (t & 7) + ((t >> 3) << 4) + 8 * hi;
      Bm[t] = (_Float16)x[(b * S_ + j0 + K) * H_ + colH];
    }

    // D(16x16,f32) += A(16x32,f16) x B(32x16,f16); uniform flow, EXEC all-1s
    acc = __builtin_amdgcn_wmma_f32_16x16x32_f16(
        false, Am.v, false, Bm, (short)0, acc, false, false);
  }

  __syncthreads();
  // C/D layout: VGPR r -> row r + 8*hi, col = lane&15 -> colH
#pragma unroll
  for (int r = 0; r < 8; ++r) {
    const int ii   = r + 8 * hi;
    const float iv = 1.0f / (sDen[ii] + 1e-10f);
    out[(b * S_ + i0 + ii) * H_ + colH] = acc[r] * iv;
  }
}

extern "C" void kernel_launch(void* const* d_in, const int* in_sizes, int n_in,
                              void* d_out, int out_size, void* d_ws, size_t ws_size,
                              hipStream_t stream) {
  const float* x  = (const float*)d_in[0];
  const float* W1 = (const float*)d_in[1];
  const float* b1 = (const float*)d_in[2];
  const float* w2 = (const float*)d_in[3];
  const float* b2 = (const float*)d_in[4];
  float* out = (float*)d_out;

  float* F = (float*)d_ws;                 // [B*S, A]
  float* G = F + (size_t)B_ * S_ * A_;     // [B*S, A] (= g + b1), ws usage: 1 MB

  fg_kernel<<<(B_ * S_ * A_) / 256, 256, 0, stream>>>(x, W1, b1, F, G);
  attn_kernel<<<B_ * (S_ / 16), 256, 0, stream>>>(x, F, G, w2, b2, out);
}
